// ODEBlock1D_73882027426314
// MI455X (gfx1250) — matrix operand order repaired and checked by reading, not compiled, not run on hardware
//
#include <hip/hip_runtime.h>

// ---------------- problem constants ----------------
#define BATCH   32
#define LLEN    8192
#define CCH     64       // channels (C_OUT)
#define NSTEPS  8
#define TILE_L  128      // output positions per block (8 waves * 16)
#define ZROWS   130      // TILE_L + 2 halo rows
#define ZSTRIDE 72       // ushorts per staged z row (64 ch + 8 pad) -> 144B, 16B aligned
#define KDIM    192      // GEMM K = 3 taps * 64 channels (t-channel folded into bias)

typedef __attribute__((ext_vector_type(16))) __bf16 v16bf;
typedef __attribute__((ext_vector_type(8)))  float  v8f;
typedef __attribute__((ext_vector_type(4)))  unsigned int v4u;
typedef __attribute__((ext_vector_type(4)))  int          v4i;
typedef __attribute__((ext_vector_type(8)))  int          v8i;

union FragBF {
    v16bf v;
    uint4 q[2];
};

__device__ __forceinline__ unsigned short f2bf(float f) {
    unsigned int u = __float_as_uint(f);
    unsigned int r = u + 0x7FFFu + ((u >> 16) & 1u);   // round-to-nearest-even
    return (unsigned short)(r >> 16);
}

// ---------------- weight prep: pack W^T as bf16 [co][kw*64+ci], t-channel sums ----------------
__global__ void __launch_bounds__(256)
ode_prep_kernel(const float* __restrict__ w,            // (3, 65, 64)
                unsigned short* __restrict__ wT,        // (64, 192) bf16
                float* __restrict__ bsum,               // sum_kw w[kw,64,co]
                float* __restrict__ wt0,                // w[0,64,co]
                float* __restrict__ wt2) {              // w[2,64,co]
    int tid = threadIdx.x;
    for (int i = tid; i < CCH * KDIM; i += 256) {
        int co = i / KDIM;
        int kl = i % KDIM;
        int kw = kl >> 6;
        int ci = kl & 63;
        wT[i] = f2bf(w[(kw * 65 + ci) * 64 + co]);
    }
    if (tid < CCH) {
        float a0 = w[(0 * 65 + 64) * 64 + tid];
        float a1 = w[(1 * 65 + 64) * 64 + tid];
        float a2 = w[(2 * 65 + 64) * 64 + tid];
        bsum[tid] = a0 + a1 + a2;
        wt0[tid]  = a0;
        wt2[tid]  = a2;
    }
}

// ---------------- state init copy ----------------
__global__ void __launch_bounds__(256)
ode_copy4_kernel(const float4* __restrict__ in, float4* __restrict__ out, int n) {
    int i = blockIdx.x * 256 + threadIdx.x;
    if (i < n) out[i] = in[i];
}

// ---------------- fused conv eval ----------------
// e = relu(conv(ybase + cmix*kprev, t) + b)
//   mode 0: acc  = e          ; kout = e
//   mode 1: acc += acc_coef*e ; kout = e
//   mode 2: ynext = ybase + step_c*(acc + e)          (RK4 state update fused into k4)
__global__ void __launch_bounds__(256)
ode_conv_kernel(const float* __restrict__ ybase,
                const float* __restrict__ kprev,   // nullptr for k1
                float cmix,
                float* __restrict__ kout,
                float* __restrict__ accb,
                float* __restrict__ ynext,
                float acc_coef, int mode, float step_c,
                float tval,
                const unsigned short* __restrict__ wTg,
                const float* __restrict__ bsum,
                const float* __restrict__ wt0,
                const float* __restrict__ wt2,
                const float* __restrict__ bias) {
    __shared__ __align__(16) unsigned short zs[ZROWS * ZSTRIDE];   // staged activations (bf16)
    __shared__ __align__(16) unsigned short wlds[CCH * KDIM];      // weights W^T (bf16)

    const int tid  = threadIdx.x;
    const int lane = tid & 31;
    const int wv   = tid >> 5;              // wave id 0..7
    const int bidx = blockIdx.x >> 6;       // batch row (LLEN/TILE_L = 64 blocks per row)
    const int l0   = (blockIdx.x & 63) * TILE_L;

    // ---- weight tile global->LDS via Tensor Data Mover (one DMA per block) ----
#if __has_builtin(__builtin_amdgcn_tensor_load_to_lds)
    if (wv == 0) {
        // D# group 0: count=1, lds_addr, global_addr[56:0], type=2
        unsigned long long ga = (unsigned long long)(size_t)wTg;
        v4u g0;
        g0.x = 1u;                                          // count = 1 valid descriptor
        g0.y = (unsigned int)(size_t)(&wlds[0]);            // LDS byte offset (addr[31:0])
        g0.z = (unsigned int)ga;                            // global_addr lo
        g0.w = ((unsigned int)(ga >> 32) & 0x01FFFFFFu)     // global_addr hi (bits 56:32)
               | 0x80000000u;                               // type = 2 ("image")
        // D# group 1: data_size=4B, 1 row x 6144 dwords, tile == tensor
        v8i g1;
        g1[0] = 0x00020000;                 // workgroup_mask=0, data_size=2 (4 bytes)
        g1[1] = (int)(6144u << 16);         // tensor_dim0[15:0] in [31:16]
        g1[2] = (int)(1u << 16);            // tensor_dim0[31:16]=0 ; tensor_dim1=1
        g1[3] = (int)(6144u << 16);         // tile_dim0 = 6144
        g1[4] = 1;                          // tile_dim1 = 1, tile_dim2 = 0
        g1[5] = 6144;                       // tensor_dim0_stride[31:0]
        g1[6] = 0;
        g1[7] = 0;
        v4i gz = {0, 0, 0, 0};
#if __clang_major__ >= 23
        v8i gz8 = {0, 0, 0, 0, 0, 0, 0, 0};
        __builtin_amdgcn_tensor_load_to_lds(g0, g1, gz, gz, gz8, 0);
#else
        __builtin_amdgcn_tensor_load_to_lds(g0, g1, gz, gz, 0);
#endif
        __builtin_amdgcn_s_wait_tensorcnt(0);
    }
#else
    {
        const unsigned int* wsrc = (const unsigned int*)wTg;
        unsigned int* wdst = (unsigned int*)wlds;
        for (int i = tid; i < (CCH * KDIM) / 2; i += 256) wdst[i] = wsrc[i];
    }
#endif

    // ---- stage z tile: rows l0-1 .. l0+128, convert (y + cmix*kprev) -> bf16 ----
    for (int i = tid; i < ZROWS * 32; i += 256) {
        int row = i >> 5;
        int pr  = i & 31;                   // pair of channels
        int gl  = l0 - 1 + row;
        unsigned int pack = 0u;
        if (gl >= 0 && gl < LLEN) {
            size_t base = ((size_t)bidx * LLEN + gl) * CCH + pr * 2;
            float2 yv = *(const float2*)(ybase + base);
            if (kprev) {
                float2 kv = *(const float2*)(kprev + base);
                yv.x = fmaf(cmix, kv.x, yv.x);
                yv.y = fmaf(cmix, kv.y, yv.y);
            }
            pack = (unsigned int)f2bf(yv.x) | ((unsigned int)f2bf(yv.y) << 16);
        }
        *(unsigned int*)&zs[row * ZSTRIDE + pr * 2] = pack;
    }
    __syncthreads();

    // ---- WMMA main loop: wave computes 16 positions x 64 channels ----
    const int m  = lane & 15;
    const int hl = lane >> 4;               // lane half
    v8f cacc[4] = {v8f{}, v8f{}, v8f{}, v8f{}};

#pragma unroll
    for (int kc = 0; kc < 6; ++kc) {
        // A fragment: 16x32 bf16.  lane<16: K 0..7 & 16..23 ; lane>=16: K 8..15 & 24..31
        const int kw   = kc >> 1;
        const int cib  = (kc & 1) * 32 + hl * 8;
        const int zrow = wv * 16 + m + kw;  // staged row = global_l - (l0-1)
        FragBF A;
        A.q[0] = *(const uint4*)&zs[zrow * ZSTRIDE + cib];
        A.q[1] = *(const uint4*)&zs[zrow * ZSTRIDE + cib + 16];

        // B fragments: 32x16 bf16.  lane<16: K 0..7 & 16..23 ; lane>=16: K 8..15 & 24..31, column = lane%16
        const int kb0 = kc * 32 + hl * 8;
#pragma unroll
        for (int nt = 0; nt < 4; ++nt) {
            const int co = nt * 16 + m;
            FragBF Bf;
            const uint4* pb = (const uint4*)&wlds[co * KDIM + kb0];
            Bf.q[0] = pb[0];
            Bf.q[1] = pb[2];
            cacc[nt] = __builtin_amdgcn_wmma_f32_16x16x32_bf16(
                false, A.v, false, Bf.v, (short)0, cacc[nt], false, false);
        }
    }

    // ---- epilogue: bias + t-channel term, edge fix, relu, RK4 bookkeeping ----
#pragma unroll
    for (int nt = 0; nt < 4; ++nt) {
        const int co = nt * 16 + m;
        const float be    = bias[co] + tval * bsum[co];
        const float bfix0 = tval * wt0[co];
        const float bfix2 = tval * wt2[co];
#pragma unroll
        for (int r = 0; r < 8; ++r) {
            const int mm = r + hl * 8;
            const int gl = l0 + wv * 16 + mm;
            float bb = be;
            if (gl == 0)        bb -= bfix0;   // left edge: kw=0 tap padded with 0 (incl. t)
            if (gl == LLEN - 1) bb -= bfix2;   // right edge: kw=2 tap padded with 0
            float e = cacc[nt][r] + bb;
            e = e > 0.0f ? e : 0.0f;
            const size_t idx = ((size_t)bidx * LLEN + gl) * CCH + co;
            if (mode == 2) {
                // y_next = y + (h/6) * (k1 + 2k2 + 2k3 + k4)
                ynext[idx] = fmaf(step_c, accb[idx] + e, ybase[idx]);
            } else {
                kout[idx] = e;
                if (mode == 0) accb[idx] = acc_coef * e;
                else           accb[idx] = fmaf(acc_coef, e, accb[idx]);
            }
        }
    }
}

// ---------------- host side ----------------
extern "C" void kernel_launch(void* const* d_in, const int* in_sizes, int n_in,
                              void* d_out, int out_size, void* d_ws, size_t ws_size,
                              hipStream_t stream) {
    (void)in_sizes; (void)n_in; (void)out_size; (void)ws_size;

    const float* x = (const float*)d_in[0];   // (32, 8192, 64) f32
    const float* w = (const float*)d_in[1];   // (3, 65, 64)    f32
    const float* b = (const float*)d_in[2];   // (64,)          f32

    char* ws = (char*)d_ws;
    unsigned short* wT = (unsigned short*)ws;           // 24576 B
    float* bsum = (float*)(ws + 24576);                 // 256 B
    float* wt0  = (float*)(ws + 24832);                 // 256 B
    float* wt2  = (float*)(ws + 25088);                 // 256 B
    const size_t NB = (size_t)BATCH * LLEN * CCH * sizeof(float);   // 64 MB
    float* accb = (float*)(ws + 65536);
    float* ka   = (float*)(ws + 65536 + NB);
    float* kb   = (float*)(ws + 65536 + 2 * NB);
    float* yB   = (float*)(ws + 65536 + 3 * NB);

    ode_prep_kernel<<<1, 256, 0, stream>>>(w, wT, bsum, wt0, wt2);

    const int n4 = BATCH * LLEN * CCH / 4;
    float* ycur = (float*)d_out;              // ping-pong: even # of steps ends in d_out
    float* ynxt = yB;
    ode_copy4_kernel<<<n4 / 256, 256, 0, stream>>>((const float4*)x, (float4*)ycur, n4);

    const float h = 1.0f / NSTEPS;
    const dim3 grid(BATCH * LLEN / TILE_L);   // 2048 blocks

    for (int s = 0; s < NSTEPS; ++s) {
        const float t0 = (float)s * h;
        // k1 = f(y, t0)                        acc  = k1
        ode_conv_kernel<<<grid, 256, 0, stream>>>(ycur, nullptr, 0.0f, ka, accb, nullptr,
                                                  1.0f, 0, 0.0f, t0,
                                                  wT, bsum, wt0, wt2, b);
        // k2 = f(y + h/2*k1, t0+h/2)           acc += 2*k2
        ode_conv_kernel<<<grid, 256, 0, stream>>>(ycur, ka, 0.5f * h, kb, accb, nullptr,
                                                  2.0f, 1, 0.0f, t0 + 0.5f * h,
                                                  wT, bsum, wt0, wt2, b);
        // k3 = f(y + h/2*k2, t0+h/2)           acc += 2*k3
        ode_conv_kernel<<<grid, 256, 0, stream>>>(ycur, kb, 0.5f * h, ka, accb, nullptr,
                                                  2.0f, 1, 0.0f, t0 + 0.5f * h,
                                                  wT, bsum, wt0, wt2, b);
        // k4 = f(y + h*k3, t0+h);  y_next = y + h/6*(acc + k4)   (fused update)
        ode_conv_kernel<<<grid, 256, 0, stream>>>(ycur, ka, h, kb, accb, ynxt,
                                                  1.0f, 2, h / 6.0f, t0 + h,
                                                  wT, bsum, wt0, wt2, b);
        float* tmp = ycur; ycur = ynxt; ynxt = tmp;
    }
}
